// TransformerBlock_66159676228026
// MI455X (gfx1250) — matrix operand order repaired and checked by reading, not compiled, not run on hardware
//
#include <hip/hip_runtime.h>
#include <hip/hip_bf16.h>

#define DI __device__ __forceinline__

constexpr int B_  = 2;
constexpr int T_  = 2048;
constexpr int C_  = 768;
constexpr int H_  = 12;
constexpr int HD_ = 64;
constexpr int BT_ = B_ * T_;
constexpr int FC_ = 4 * C_;   // 3072

typedef __attribute__((ext_vector_type(16))) __bf16 v16bf;
typedef __attribute__((ext_vector_type(8)))  float  v8f;
typedef __attribute__((ext_vector_type(4)))  unsigned int u32x4;
typedef __attribute__((ext_vector_type(8)))  int i32x8;
typedef __attribute__((ext_vector_type(4)))  int i32x4;

struct alignas(16) U4 { unsigned int x, y, z, w; };

union FragBF {
    v16bf v;
    U4 q[2];
    unsigned short u[16];
};

union BQU { U4 q; unsigned short u[8]; };

DI unsigned short f32_bf16(float f) {
    unsigned int u = __float_as_uint(f);
    u += 0x7FFFu + ((u >> 16) & 1u);     // round-to-nearest-even
    return (unsigned short)(u >> 16);
}

DI U4 ldg16(const unsigned short* p) { return *reinterpret_cast<const U4*>(p); }

DI v8f wmma_bf16(const FragBF& a, const FragBF& b, v8f c) {
    return __builtin_amdgcn_wmma_f32_16x16x32_bf16(
        /*neg_a=*/false, a.v, /*neg_b=*/false, b.v,
        /*c_mod=*/(short)0, c, /*reuse_a=*/false, /*reuse_b=*/false);
}

// ---- CDNA5 async data-path selection ---------------------------------------
#if __has_builtin(__builtin_amdgcn_tensor_load_to_lds) && __has_builtin(__builtin_amdgcn_s_wait_tensorcnt)
#define USE_TDM 1
#elif __has_builtin(__builtin_amdgcn_global_load_async_to_lds_b128) && __has_builtin(__builtin_amdgcn_s_wait_asynccnt)
#define USE_ASYNC 1
#endif

#if defined(USE_TDM)
// Issue one TDM 2D tile load: tile_h rows x tile_w elems (bf16) from
// row-major global (row stride = stride_elems) into contiguous LDS.
DI void tdm_load_tile(unsigned lds_byte_addr, const unsigned short* gp,
                      int tile_w, int tile_h, int stride_elems, int rows_total)
{
    const unsigned long long ga = (unsigned long long)(size_t)gp;
    u32x4 g0;
    g0.x = 1u;                                        // count=1, user descriptor
    g0.y = lds_byte_addr;                             // lds_addr
    g0.z = (unsigned)(ga & 0xFFFFFFFFull);            // global_addr[31:0]
    g0.w = (unsigned)((ga >> 32) & 0x01FFFFFFull)     // global_addr[56:32]
         | (2u << 30);                                // type = 2 ("image")
    const unsigned td0 = (unsigned)stride_elems;      // tensor_dim0 (elements)
    const unsigned td1 = (unsigned)rows_total;        // tensor_dim1
    i32x8 g1;
    g1[0] = (int)(1u << 16);                          // mask=0, data_size=1 (2B)
    g1[1] = (int)((td0 & 0xFFFFu) << 16);             // tensor_dim0[15:0]
    g1[2] = (int)((td0 >> 16) | ((td1 & 0xFFFFu) << 16));
    g1[3] = (int)((td1 >> 16) | ((unsigned)tile_w << 16));   // tile_dim0
    g1[4] = (int)((unsigned)tile_h & 0xFFFFu);        // tile_dim1, tile_dim2=0
    g1[5] = (int)(unsigned)stride_elems;              // tensor_dim0_stride[31:0]
    g1[6] = 0;                                        // stride hi, dim1_stride lo
    g1[7] = 0;
    i32x4 z4 = {0, 0, 0, 0};
#if __clang_major__ >= 23
    i32x8 z8 = {0, 0, 0, 0, 0, 0, 0, 0};
    __builtin_amdgcn_tensor_load_to_lds(g0, g1, z4, z4, z8, 0);
#else
    __builtin_amdgcn_tensor_load_to_lds(g0, g1, z4, z4, 0);
#endif
}
#endif

// ---------------------------------------------------------------- fp32->bf16
__global__ __launch_bounds__(256) void cvt_f32_bf16_kernel(
    const float* __restrict__ in, unsigned short* __restrict__ out, int n)
{
    int i = blockIdx.x * blockDim.x + threadIdx.x;
    int stride = gridDim.x * blockDim.x;
    for (; i < n; i += stride) out[i] = f32_bf16(in[i]);
}

// ---------------------------------------------------------------- LayerNorm
__global__ __launch_bounds__(256) void ln_bf16_kernel(
    const float* __restrict__ x, const float* __restrict__ gam,
    const float* __restrict__ bet, unsigned short* __restrict__ out)
{
    __shared__ float rs[256], rq[256];
    const int row = blockIdx.x, tid = threadIdx.x;
    const float* xr = x + (size_t)row * C_;
    float a0 = xr[tid], a1 = xr[tid + 256], a2 = xr[tid + 512];
    rs[tid] = a0 + a1 + a2;
    rq[tid] = a0 * a0 + a1 * a1 + a2 * a2;
    __syncthreads();
    for (int o = 128; o > 0; o >>= 1) {
        if (tid < o) { rs[tid] += rs[tid + o]; rq[tid] += rq[tid + o]; }
        __syncthreads();
    }
    const float mu  = rs[0] * (1.f / C_);
    const float var = rq[0] * (1.f / C_) - mu * mu;
    const float inv = rsqrtf(var + 1e-5f);
    unsigned short* orow = out + (size_t)row * C_;
    orow[tid]       = f32_bf16(gam[tid]       * (a0 - mu) * inv + bet[tid]);
    orow[tid + 256] = f32_bf16(gam[tid + 256] * (a1 - mu) * inv + bet[tid + 256]);
    orow[tid + 512] = f32_bf16(gam[tid + 512] * (a2 - mu) * inv + bet[tid + 512]);
}

// ------------------------------------------------------------- tile helpers
DI void load_a_regs(const unsigned short* A, int blockM, int kb, int K,
                    int tid, U4 aq[2]) {
#pragma unroll
    for (int p = 0; p < 2; ++p) {
        int idx = tid + p * 256;
        int row = idx >> 2, kc = (idx & 3) << 3;
        aq[p] = ldg16(A + (size_t)(blockM + row) * K + kb + kc);
    }
}
DI void store_a_lds(unsigned short* lA, int tid, const U4 aq[2]) {
#pragma unroll
    for (int p = 0; p < 2; ++p) {
        int idx = tid + p * 256;
        int row = idx >> 2, kc = (idx & 3) << 3;
        *reinterpret_cast<U4*>(&lA[row * 32 + kc]) = aq[p];
    }
}
DI void load_b_regs(const unsigned short* Bw, int blockN, int kb, int N,
                    int tid, BQU bq[2]) {
#pragma unroll
    for (int p = 0; p < 2; ++p) {
        int idx = tid + p * 256;
        int kk = idx >> 4, nn = (idx & 15) << 3;
        bq[p].q = ldg16(Bw + (size_t)(kb + kk) * N + blockN + nn);
    }
}
DI void store_b_lds(unsigned short* lB, int tid, const BQU bq[2]) {
#pragma unroll
    for (int p = 0; p < 2; ++p) {
        int idx = tid + p * 256;
        int kk = idx >> 4, nn = (idx & 15) << 3;
#pragma unroll
        for (int j = 0; j < 8; ++j) lB[(nn + j) * 32 + kk] = bq[p].u[j];
    }
}

// ---------------------------------------------------------------- bf16 GEMM
// C[M,N] = A[M,K] @ B[K,N] (+bias)(gelu)(+res). 128x128 block tile, 8 waves
// (4 in M x 2 in N); double-buffered LDS; A tile moved by the TDM.
template<bool GELU, bool RES, bool VTRANS>
__global__ __launch_bounds__(256) void gemm_bf16_kernel(
    const unsigned short* __restrict__ A, const unsigned short* __restrict__ Bw,
    const float* __restrict__ bias, const float* __restrict__ res,
    float* __restrict__ outF, unsigned short* __restrict__ outB,
    int Mdim, int N, int K)
{
    __shared__ unsigned short lA[2][128 * 32];   // [m][k]
    __shared__ unsigned short lB[2][128 * 32];   // [n][k] (transposed)

    const int tid   = threadIdx.x;
    const int wave  = tid >> 5, lane = tid & 31;
    const int r     = lane & 15, g = lane >> 4;
    const int waveM = wave & 3,  waveN = wave >> 2;
    const int blockM = blockIdx.x * 128, blockN = blockIdx.y * 128;

    v8f acc[2][4];
#pragma unroll
    for (int i = 0; i < 2; ++i)
#pragma unroll
        for (int j = 0; j < 4; ++j)
#pragma unroll
            for (int e = 0; e < 8; ++e) acc[i][j][e] = 0.f;

    // ---- prologue: fill buffer 0
    {
#if defined(USE_TDM)
        if (tid < 32)
            tdm_load_tile((unsigned)(size_t)&lA[0][0],
                          A + (size_t)blockM * K, 32, 128, K, Mdim);
#elif defined(USE_ASYNC)
#pragma unroll
        for (int p = 0; p < 2; ++p) {
            int idx = tid + p * 256;
            int row = idx >> 2, kc = (idx & 3) << 3;
            __builtin_amdgcn_global_load_async_to_lds_b128(
                (__attribute__((address_space(1))) void*)(void*)
                    (A + (size_t)(blockM + row) * K + kc),
                (__attribute__((address_space(3))) void*)(void*)
                    &lA[0][row * 32 + kc], 0, 0);
        }
#else
        U4 aq[2];
        load_a_regs(A, blockM, 0, K, tid, aq);
        store_a_lds(lA[0], tid, aq);
#endif
        BQU bq[2];
        load_b_regs(Bw, blockN, 0, N, tid, bq);
        store_b_lds(lB[0], tid, bq);
#if defined(USE_TDM)
        if (tid < 32) __builtin_amdgcn_s_wait_tensorcnt(0);
#elif defined(USE_ASYNC)
        __builtin_amdgcn_s_wait_asynccnt(0);
#endif
        __syncthreads();
    }

    const int nIter = K >> 5;
    for (int it = 0; it < nIter; ++it) {
        const int cur = it & 1, nxt = cur ^ 1;
        const int kbn = (it + 1) << 5;
        const bool hasNext = (it + 1) < nIter;

        BQU bq[2];
#if !defined(USE_TDM) && !defined(USE_ASYNC)
        U4 aq[2];
#endif
        if (hasNext) {
#if defined(USE_TDM)
            if (tid < 32)
                tdm_load_tile((unsigned)(size_t)&lA[nxt][0],
                              A + (size_t)blockM * K + kbn, 32, 128, K, Mdim);
#elif defined(USE_ASYNC)
#pragma unroll
            for (int p = 0; p < 2; ++p) {
                int idx = tid + p * 256;
                int row = idx >> 2, kc = (idx & 3) << 3;
                __builtin_amdgcn_global_load_async_to_lds_b128(
                    (__attribute__((address_space(1))) void*)(void*)
                        (A + (size_t)(blockM + row) * K + kbn + kc),
                    (__attribute__((address_space(3))) void*)(void*)
                        &lA[nxt][row * 32 + kc], 0, 0);
            }
#else
            load_a_regs(A, blockM, kbn, K, tid, aq);
#endif
            load_b_regs(Bw, blockN, kbn, N, tid, bq);
            __builtin_prefetch(Bw + (size_t)(kbn + (tid >> 4)) * N + blockN, 0, 1);
        }

        // fragments from current buffer (layouts per CDNA5 ISA 7.12.2)
        FragBF aF[2], bF[4];
#pragma unroll
        for (int mt = 0; mt < 2; ++mt) {
            int rb = (waveM * 32 + mt * 16 + r) * 32;
            aF[mt].q[0] = *reinterpret_cast<const U4*>(&lA[cur][rb + g * 8]);
            aF[mt].q[1] = *reinterpret_cast<const U4*>(&lA[cur][rb + 16 + g * 8]);
        }
#pragma unroll
        for (int nt = 0; nt < 4; ++nt) {
            int nb = (waveN * 64 + nt * 16 + r) * 32;
            bF[nt].q[0] = *reinterpret_cast<const U4*>(&lB[cur][nb + g * 8]);
            bF[nt].q[1] = *reinterpret_cast<const U4*>(&lB[cur][nb + 16 + g * 8]);
        }
#pragma unroll
        for (int mt = 0; mt < 2; ++mt)
#pragma unroll
            for (int nt = 0; nt < 4; ++nt)
                acc[mt][nt] = wmma_bf16(aF[mt], bF[nt], acc[mt][nt]);

        if (hasNext) {
            store_b_lds(lB[nxt], tid, bq);
#if defined(USE_TDM)
            if (tid < 32) __builtin_amdgcn_s_wait_tensorcnt(0);
#elif defined(USE_ASYNC)
            __builtin_amdgcn_s_wait_asynccnt(0);
#else
            store_a_lds(lA[nxt], tid, aq);
#endif
            __syncthreads();   // publish next buffers (one barrier per K-step)
        }
    }

    // epilogue: lane holds col = n (lane%16), rows vv + 8*(lane/16)
#pragma unroll
    for (int mt = 0; mt < 2; ++mt)
#pragma unroll
        for (int nt = 0; nt < 4; ++nt) {
            const int col = blockN + waveN * 64 + nt * 16 + r;
            const float bsv = bias ? bias[col] : 0.f;
#pragma unroll
            for (int vv = 0; vv < 8; ++vv) {
                const int row = blockM + waveM * 32 + mt * 16 + vv + 8 * g;
                float xv = acc[mt][nt][vv] + bsv;
                if (GELU) xv = 0.5f * xv * (1.f + erff(xv * 0.70710678118654752f));
                if (RES)  xv += res[(size_t)row * N + col];
                if (outF) outF[(size_t)row * N + col] = xv;
                if (outB) {
                    if (VTRANS) {
                        int bb = row / T_, t = row - bb * T_;
                        int hh = col / HD_, d = col - hh * HD_;
                        outB[(((size_t)bb * H_ + hh) * HD_ + d) * T_ + t] = f32_bf16(xv);
                    } else {
                        outB[(size_t)row * N + col] = f32_bf16(xv);
                    }
                }
            }
        }
}

// ---------------------------------------------------------------- attention
// flash attention, one wave per 16 query rows of one (b,h).
// q,k in [B,T,C] bf16; vt in [B,H,HD,T] bf16; ctx out [B,T,C] bf16.
__global__ __launch_bounds__(256) void attn_kernel(
    const unsigned short* __restrict__ q, const unsigned short* __restrict__ k,
    const unsigned short* __restrict__ vt, unsigned short* __restrict__ ctx)
{
    __shared__ unsigned short pT[8][16 * 32];   // per-wave P transpose bounce

    const int wave = threadIdx.x >> 5, lane = threadIdx.x & 31;
    const int r = lane & 15, g = lane >> 4;
    const int task = blockIdx.x * 8 + wave;     // 0 .. B*H*(T/16)-1
    const int qt = task & (T_ / 16 - 1);
    const int bh = task >> 7;                   // T/16 == 128
    const int h  = bh % H_, b = bh / H_;
    const int q0 = qt * 16;
    const float scale = 0.125f;                 // 1/sqrt(64)

    const unsigned short* qrow = q + ((size_t)(b * T_ + q0 + r)) * C_ + h * HD_;
    FragBF qA[2];
    qA[0].q[0] = ldg16(qrow + g * 8);
    qA[0].q[1] = ldg16(qrow + 16 + g * 8);
    qA[1].q[0] = ldg16(qrow + 32 + g * 8);
    qA[1].q[1] = ldg16(qrow + 48 + g * 8);

    v8f accd[4];
#pragma unroll
    for (int dt = 0; dt < 4; ++dt)
#pragma unroll
        for (int e = 0; e < 8; ++e) accd[dt][e] = 0.f;
    float mrow[8], lrow[8];
#pragma unroll
    for (int vv = 0; vv < 8; ++vv) { mrow[vv] = -3.0e38f; lrow[vv] = 0.f; }

    const int nk = q0 + 16;                     // causal: keys 0..q0+15
    for (int key0 = 0; key0 < nk; key0 += 32) {
        v8f s0, s1;
#pragma unroll
        for (int e = 0; e < 8; ++e) { s0[e] = 0.f; s1[e] = 0.f; }
#pragma unroll
        for (int kb2 = 0; kb2 < 2; ++kb2) {
            const unsigned short* k0p =
                k + ((size_t)(b * T_ + key0 + r)) * C_ + h * HD_ + kb2 * 32;
            const unsigned short* k1p =
                k + ((size_t)(b * T_ + key0 + 16 + r)) * C_ + h * HD_ + kb2 * 32;
            FragBF kf0, kf1;
            kf0.q[0] = ldg16(k0p + g * 8); kf0.q[1] = ldg16(k0p + 16 + g * 8);
            kf1.q[0] = ldg16(k1p + g * 8); kf1.q[1] = ldg16(k1p + 16 + g * 8);
            s0 = wmma_bf16(qA[kb2], kf0, s0);
            s1 = wmma_bf16(qA[kb2], kf1, s1);
        }
#pragma unroll
        for (int vv = 0; vv < 8; ++vv) {
            const int qg = q0 + vv + 8 * g;
            float t0 = s0[vv] * scale; if (key0 + r      > qg) t0 = -1.0e30f;
            float t1 = s1[vv] * scale; if (key0 + 16 + r > qg) t1 = -1.0e30f;
            float mx = fmaxf(t0, t1);
#pragma unroll
            for (int off = 1; off < 16; off <<= 1)
                mx = fmaxf(mx, __shfl_xor(mx, off, 32));
            const float mn = fmaxf(mrow[vv], mx);
            const float corr = __expf(mrow[vv] - mn);
            const float p0 = __expf(t0 - mn);
            const float p1 = __expf(t1 - mn);
            float rsum = p0 + p1;
#pragma unroll
            for (int off = 1; off < 16; off <<= 1)
                rsum += __shfl_xor(rsum, off, 32);
            lrow[vv] = lrow[vv] * corr + rsum;
            mrow[vv] = mn;
#pragma unroll
            for (int dt = 0; dt < 4; ++dt) accd[dt][vv] *= corr;
            pT[wave][(vv + 8 * g) * 32 + r]      = f32_bf16(p0);
            pT[wave][(vv + 8 * g) * 32 + 16 + r] = f32_bf16(p1);
        }
        FragBF pf;
        pf.q[0] = *reinterpret_cast<const U4*>(&pT[wave][r * 32 + g * 8]);
        pf.q[1] = *reinterpret_cast<const U4*>(&pT[wave][r * 32 + 16 + g * 8]);
#pragma unroll
        for (int dt = 0; dt < 4; ++dt) {
            const unsigned short* vp =
                vt + ((size_t)(b * H_ + h) * HD_ + dt * 16 + r) * T_ + key0;
            FragBF vf;
            vf.q[0] = ldg16(vp + g * 8); vf.q[1] = ldg16(vp + 16 + g * 8);
            accd[dt] = wmma_bf16(pf, vf, accd[dt]);
        }
    }

#pragma unroll
    for (int dt = 0; dt < 4; ++dt)
#pragma unroll
        for (int vv = 0; vv < 8; ++vv) {
            const float val = accd[dt][vv] / lrow[vv];
            const int row = b * T_ + q0 + vv + 8 * g;
            const int col = h * HD_ + dt * 16 + r;
            ctx[(size_t)row * C_ + col] = f32_bf16(val);
        }
}

// ---------------------------------------------------------------- launcher
extern "C" void kernel_launch(void* const* d_in, const int* in_sizes, int n_in,
                              void* d_out, int out_size, void* d_ws, size_t ws_size,
                              hipStream_t stream)
{
    (void)in_sizes; (void)n_in; (void)out_size; (void)ws_size;

    const float* x      = (const float*)d_in[0];
    const float* Wq     = (const float*)d_in[1];
    const float* Wk     = (const float*)d_in[2];
    const float* Wv     = (const float*)d_in[3];
    const float* Wo     = (const float*)d_in[4];
    const float* bo     = (const float*)d_in[5];
    const float* W_fc   = (const float*)d_in[6];
    const float* b_fc   = (const float*)d_in[7];
    const float* W_proj = (const float*)d_in[8];
    const float* b_proj = (const float*)d_in[9];
    const float* g1     = (const float*)d_in[10];
    const float* be1    = (const float*)d_in[11];
    const float* g2     = (const float*)d_in[12];
    const float* be2    = (const float*)d_in[13];

    char* w = (char*)d_ws;
    auto take = [&](size_t bytes) -> void* {
        void* p = (void*)w;
        w += (bytes + 255) & ~(size_t)255;
        return p;
    };
    unsigned short* wqB    = (unsigned short*)take((size_t)C_ * C_ * 2);
    unsigned short* wkB    = (unsigned short*)take((size_t)C_ * C_ * 2);
    unsigned short* wvB    = (unsigned short*)take((size_t)C_ * C_ * 2);
    unsigned short* woB    = (unsigned short*)take((size_t)C_ * C_ * 2);
    unsigned short* wfcB   = (unsigned short*)take((size_t)C_ * FC_ * 2);
    unsigned short* wprojB = (unsigned short*)take((size_t)FC_ * C_ * 2);
    unsigned short* xn     = (unsigned short*)take((size_t)BT_ * C_ * 2);
    unsigned short* qB     = (unsigned short*)take((size_t)BT_ * C_ * 2);
    unsigned short* kB     = (unsigned short*)take((size_t)BT_ * C_ * 2);
    unsigned short* vtB    = (unsigned short*)take((size_t)BT_ * C_ * 2);
    unsigned short* ctxB   = (unsigned short*)take((size_t)BT_ * C_ * 2);
    float*          xa     = (float*)take((size_t)BT_ * C_ * 4);
    unsigned short* hB     = (unsigned short*)take((size_t)BT_ * FC_ * 2);

    cvt_f32_bf16_kernel<<<512, 256, 0, stream>>>(Wq, wqB, C_ * C_);
    cvt_f32_bf16_kernel<<<512, 256, 0, stream>>>(Wk, wkB, C_ * C_);
    cvt_f32_bf16_kernel<<<512, 256, 0, stream>>>(Wv, wvB, C_ * C_);
    cvt_f32_bf16_kernel<<<512, 256, 0, stream>>>(Wo, woB, C_ * C_);
    cvt_f32_bf16_kernel<<<1024, 256, 0, stream>>>(W_fc, wfcB, C_ * FC_);
    cvt_f32_bf16_kernel<<<1024, 256, 0, stream>>>(W_proj, wprojB, FC_ * C_);

    ln_bf16_kernel<<<BT_, 256, 0, stream>>>(x, g1, be1, xn);

    const dim3 g768(BT_ / 128, C_ / 128);     // 32 x 6
    const dim3 gfc(BT_ / 128, FC_ / 128);     // 32 x 24

    gemm_bf16_kernel<false, false, false><<<g768, 256, 0, stream>>>(
        xn, wqB, nullptr, nullptr, nullptr, qB, BT_, C_, C_);
    gemm_bf16_kernel<false, false, false><<<g768, 256, 0, stream>>>(
        xn, wkB, nullptr, nullptr, nullptr, kB, BT_, C_, C_);
    gemm_bf16_kernel<false, false, true><<<g768, 256, 0, stream>>>(
        xn, wvB, nullptr, nullptr, nullptr, vtB, BT_, C_, C_);

    attn_kernel<<<(B_ * H_ * (T_ / 16)) / 8, 256, 0, stream>>>(qB, kB, vtB, ctxB);

    gemm_bf16_kernel<false, true, false><<<g768, 256, 0, stream>>>(
        ctxB, woB, bo, x, xa, nullptr, BT_, C_, C_);

    ln_bf16_kernel<<<BT_, 256, 0, stream>>>(xa, g2, be2, xn);

    gemm_bf16_kernel<true, false, false><<<gfc, 256, 0, stream>>>(
        xn, wfcB, b_fc, nullptr, nullptr, hB, BT_, FC_, C_);

    gemm_bf16_kernel<false, true, false><<<g768, 256, 0, stream>>>(
        hB, wprojB, b_proj, xa, (float*)d_out, nullptr, BT_, C_, FC_);
}
